// SMorphLayer_46239617909552
// MI455X (gfx1250) — matrix operand order repaired and checked by reading, not compiled, not run on hardware
//
#include <hip/hip_runtime.h>
#include <hip/hip_bf16.h>

typedef __attribute__((ext_vector_type(16))) _Float16 v16h;
typedef __attribute__((ext_vector_type(8)))  _Float16 v8h;
typedef __attribute__((ext_vector_type(8)))  float    v8f;

#define BATCH 8
#define HH 128
#define WW 128
#define OH 124
#define OWID 124
#define NF 16
#define NPIX (BATCH * HH * WW)     // 131072 per plane
#define BMAT_ELEMS 8192            // 4 Ktiles * 4 Ntiles * 32 lanes * 16 elems

// ---------------------------------------------------------------------------
// Kernel 1: per-pixel exponential planes.
//   p0 = x*e^x, p1 = e^x, p2 = -x*e^-x, p3 = e^-x   (f16)
// ---------------------------------------------------------------------------
__global__ void prep_planes(const float* __restrict__ x,
                            _Float16* __restrict__ planes) {
    int i = blockIdx.x * blockDim.x + threadIdx.x;
    if (i >= NPIX) return;
    float v  = x[i];
    float e  = __expf(v);
    float en = __expf(-v);
    planes[0 * NPIX + i] = (_Float16)(v * e);
    planes[1 * NPIX + i] = (_Float16)(e);
    planes[2 * NPIX + i] = (_Float16)(-v * en);
    planes[3 * NPIX + i] = (_Float16)(en);
}

// ---------------------------------------------------------------------------
// Kernel 2: build the B matrix (K=128 x N=64) directly in per-lane WMMA
// B-fragment layout. Flat index = ((ktile*4 + ntile)*32 + lane)*16 + elem,
// with K_local = 16*(lane/16) + elem, col f = lane%16.
// K-tile t == plane t; j = K_local (valid taps j<25, zero otherwise).
//   ntile 0 (num1): t=0 -> e^{k1}, t=1 -> k1*e^{k1}
//   ntile 1 (den1): t=1 -> e^{k1}
//   ntile 2 (num2): t=2 -> e^{k2}, t=3 -> k2*e^{k2}
//   ntile 3 (den2): t=3 -> e^{k2}
// ---------------------------------------------------------------------------
__global__ void prep_weights(const float* __restrict__ k1,
                             const float* __restrict__ k2,
                             _Float16* __restrict__ Bm) {
    int idx = blockIdx.x * blockDim.x + threadIdx.x;
    if (idx >= BMAT_ELEMS) return;
    int e = idx & 15;
    int l = (idx >> 4) & 31;
    int n = (idx >> 9) & 3;
    int t = idx >> 11;
    int j = 16 * (l >> 4) + e;   // K within tile (0..31)
    int f = l & 15;
    float val = 0.0f;
    if (j < 25) {
        if (n == 0) {
            if (t == 0)      val = __expf(k1[f * 25 + j]);
            else if (t == 1) { float kv = k1[f * 25 + j]; val = kv * __expf(kv); }
        } else if (n == 1) {
            if (t == 1)      val = __expf(k1[f * 25 + j]);
        } else if (n == 2) {
            if (t == 2)      val = __expf(k2[f * 25 + j]);
            else if (t == 3) { float kv = k2[f * 25 + j]; val = kv * __expf(kv); }
        } else {
            if (t == 3)      val = __expf(k2[f * 25 + j]);
        }
    }
    Bm[idx] = (_Float16)val;
}

// ---------------------------------------------------------------------------
// Kernel 3: fused im2col GEMM + smooth-max epilogue.
// Block = 128 threads (4 wave32s). Wave w handles ow tile (blockIdx.x*4+w)*16
// at row oh=blockIdx.y, batch b=blockIdx.z. Each wave: 16 rows(M) x 64 cols(N)
// x 128 K -> 16 v_wmma_f32_16x16x32_f16.
// ---------------------------------------------------------------------------
__global__ __launch_bounds__(128) void smorph_gemm(
        const _Float16* __restrict__ planes,
        const v16h*     __restrict__ Bfrag,   // per-lane fragments (kernel 2)
        const float*    __restrict__ bias,
        float*          __restrict__ out) {
    __shared__ __align__(128) _Float16 As[4][16][128];   // [wave][M row][K]

    const int lane = threadIdx.x & 31;
    const int w    = threadIdx.x >> 5;
    const int owb  = (blockIdx.x * 4 + w) * 16;
    const int oh   = blockIdx.y;
    const int b    = blockIdx.z;
    const int r    = lane & 15;   // M row this lane stages / reads
    const int g    = lane >> 4;   // half-wave id

    // ---- stage im2col A tile: A[r][32*p + 5*dy + dx] = plane_p[oh+dy][ow+dx]
    // lane covers planes {g, g+2}; taps j=25..31 zero-padded.
    #pragma unroll
    for (int pi = 0; pi < 2; ++pi) {
        const int p = g + 2 * pi;
        const _Float16* src = planes + (p * BATCH + b) * (HH * WW);
        #pragma unroll
        for (int dy = 0; dy < 5; ++dy) {
            const _Float16* row = src + (oh + dy) * WW;
            #pragma unroll
            for (int dx = 0; dx < 5; ++dx) {
                int wi = owb + r + dx;          // <= 131 on the last tile
                if (wi > WW - 1) wi = WW - 1;   // clamp (rows masked on store)
                As[w][r][32 * p + 5 * dy + dx] = row[wi];
            }
        }
        #pragma unroll
        for (int j = 25; j < 32; ++j)
            As[w][r][32 * p + j] = (_Float16)0.0f;
    }
    __syncthreads();

    // ---- GEMM: 4 K-tiles (one plane each) x 4 N-tiles (num1,den1,num2,den2)
    v8f acc0 = {}, acc1 = {}, acc2 = {}, acc3 = {};
    #pragma unroll
    for (int t = 0; t < 4; ++t) {
        // A fragment (contiguous K order, same as Bfrag): elems 0-7 = K 16g..16g+7,
        // elems 8-15 = K 16g+8..16g+15, row = lane%16.
        const v8h lo = *(const v8h*)&As[w][r][t * 32 + 16 * g];
        const v8h hi = *(const v8h*)&As[w][r][t * 32 + 16 * g + 8];
        v16h a;
        #pragma unroll
        for (int i = 0; i < 8; ++i) { a[i] = lo[i]; a[i + 8] = hi[i]; }

        const v16h b0 = Bfrag[(t * 4 + 0) * 32 + lane];
        const v16h b1 = Bfrag[(t * 4 + 1) * 32 + lane];
        const v16h b2 = Bfrag[(t * 4 + 2) * 32 + lane];
        const v16h b3 = Bfrag[(t * 4 + 3) * 32 + lane];

        acc0 = __builtin_amdgcn_wmma_f32_16x16x32_f16(false, a, false, b0,
                                                      (short)0, acc0, false, false);
        acc1 = __builtin_amdgcn_wmma_f32_16x16x32_f16(false, a, false, b1,
                                                      (short)0, acc1, false, false);
        acc2 = __builtin_amdgcn_wmma_f32_16x16x32_f16(false, a, false, b2,
                                                      (short)0, acc2, false, false);
        acc3 = __builtin_amdgcn_wmma_f32_16x16x32_f16(false, a, false, b3,
                                                      (short)0, acc3, false, false);
    }

    // ---- epilogue: all 4 accumulator tiles share the lane<->(M,N) mapping:
    // elem v -> M = v + 8*(lane/16), N = lane%16.
    const int f = r;
    const float bv = bias[f];
    float* obase = out + ((b * NF + f) * OH + oh) * OWID;
    #pragma unroll
    for (int v = 0; v < 8; ++v) {
        const int ow = owb + v + 8 * g;
        if (ow < OWID) {
            obase[ow] = acc0[v] / acc1[v] + acc2[v] / acc3[v] + bv;
        }
    }
}

extern "C" void kernel_launch(void* const* d_in, const int* in_sizes, int n_in,
                              void* d_out, int out_size, void* d_ws, size_t ws_size,
                              hipStream_t stream) {
    const float* x    = (const float*)d_in[0];
    const float* k1   = (const float*)d_in[1];
    const float* k2   = (const float*)d_in[2];
    const float* bias = (const float*)d_in[3];
    float* out = (float*)d_out;

    _Float16* planes = (_Float16*)d_ws;                           // 4*NPIX f16 = 1 MB
    _Float16* Bm     = (_Float16*)((char*)d_ws + (1u << 20));     // 16 KB

    prep_planes<<<(NPIX + 255) / 256, 256, 0, stream>>>(x, planes);
    prep_weights<<<(BMAT_ELEMS + 255) / 256, 256, 0, stream>>>(k1, k2, Bm);
    smorph_gemm<<<dim3(2, OH, BATCH), 128, 0, stream>>>(planes, (const v16h*)Bm,
                                                        bias, out);
}
